// FP8Linear_66606352826376
// MI455X (gfx1250) — 8-bit matrix operand order repaired and checked by reading, compiled, not run on hardware
//
#include <hip/hip_runtime.h>
#include <stdint.h>

typedef __attribute__((ext_vector_type(16))) int   v16i;
typedef __attribute__((ext_vector_type(4)))  int   v4i;
typedef __attribute__((ext_vector_type(2)))  int   v2i;
typedef __attribute__((ext_vector_type(8)))  float v8f;
typedef __attribute__((ext_vector_type(4)))  float v4f;

#define QBLOCK 128
#define FP8_MAX 448.0f

__global__ void quant_x_kernel(const float* __restrict__ x,
                               uint8_t* __restrict__ xq,
                               float* __restrict__ xs_t,
                               int M, int K) {
  const int KB    = K >> 7;
  const int gwave = (int)((blockIdx.x * blockDim.x + threadIdx.x) >> 5);
  const int lane  = threadIdx.x & 31;
  const int row   = gwave / KB;
  const int kb    = gwave - row * KB;
  if (row >= M) return;

  const size_t base = (size_t)row * K + (size_t)kb * QBLOCK + (size_t)lane * 4;
  v4f v = *(const v4f*)(x + base);
  float amax = fmaxf(fmaxf(__builtin_fabsf(v.x), __builtin_fabsf(v.y)),
                     fmaxf(__builtin_fabsf(v.z), __builtin_fabsf(v.w)));
#pragma unroll
  for (int off = 16; off > 0; off >>= 1)
    amax = fmaxf(amax, __shfl_xor(amax, off, 32));

  const float s   = fmaxf(amax / FP8_MAX, 1e-12f);
  const float inv = 1.0f / s;
  int packed = 0;
  packed = __builtin_amdgcn_cvt_pk_fp8_f32(v.x * inv, v.y * inv, packed, false);
  packed = __builtin_amdgcn_cvt_pk_fp8_f32(v.z * inv, v.w * inv, packed, true);
  *(int*)(xq + base) = packed;
  if (lane == 0) xs_t[(size_t)kb * M + row] = s;
}

__global__ void quant_w_kernel(const float* __restrict__ w,
                               uint8_t* __restrict__ wq, long n4) {
  const long i = (long)blockIdx.x * blockDim.x + threadIdx.x;
  if (i >= n4) return;
  v4f v = *(const v4f*)(w + i * 4);
  int p = 0;
  p = __builtin_amdgcn_cvt_pk_fp8_f32(v.x, v.y, p, false);
  p = __builtin_amdgcn_cvt_pk_fp8_f32(v.z, v.w, p, true);
  ((int*)wq)[i] = p;
}

__global__ void __launch_bounds__(256)
fp8_gemm_kernel(const uint8_t* __restrict__ xq,
                const float* __restrict__ xs_t,
                const uint8_t* __restrict__ wq,
                const float* __restrict__ wscale,
                const float* __restrict__ bias,
                float* __restrict__ y,
                int M, int N, int K) {
  const int lane = threadIdx.x & 31;
  const int wave = threadIdx.x >> 5;
  const int wm   = wave >> 2;
  const int wn   = wave & 3;
  const int h    = lane >> 4;
  const int l16  = lane & 15;
  const int Mbase = blockIdx.y * 64 + wm * 32;
  const int Nbase = blockIdx.x * 256 + wn * 64;
  const int KB = K >> 7;

  v8f acc[2][4];
#pragma unroll
  for (int mt = 0; mt < 2; ++mt)
#pragma unroll
    for (int nt = 0; nt < 4; ++nt)
      acc[mt][nt] = v8f{};

  const uint8_t* aBase = xq + (size_t)(Mbase + l16) * K + (size_t)h * 16;

  for (int kb = 0; kb < KB; ++kb) {
    const int k0 = kb * QBLOCK;

    v16i Bt[4];
#pragma unroll
    for (int nt = 0; nt < 4; ++nt) {
      const uint8_t* bp =
          wq + (size_t)(Nbase + nt * 16 + l16) * K + k0 + (size_t)h * 16;
#pragma unroll
      for (int g = 0; g < 4; ++g) {
        v4i t = *(const v4i*)(bp + g * 32);
        Bt[nt][4 * g + 0] = t.x;
        Bt[nt][4 * g + 1] = t.y;
        Bt[nt][4 * g + 2] = t.z;
        Bt[nt][4 * g + 3] = t.w;
      }
    }

    v4f sc[2][2];
#pragma unroll
    for (int mt = 0; mt < 2; ++mt) {
      const float* sp = xs_t + (size_t)kb * M + Mbase + mt * 16 + h * 8;
      sc[mt][0] = *(const v4f*)sp;
      sc[mt][1] = *(const v4f*)(sp + 4);
    }

#pragma unroll
    for (int mt = 0; mt < 2; ++mt) {
      v16i At;
      const uint8_t* ap = aBase + (size_t)(mt * 16) * K + k0;
#pragma unroll
      for (int s = 0; s < 2; ++s)
#pragma unroll
        for (int j = 0; j < 4; ++j) {
          v2i t = *(const v2i*)(ap + s * 64 + (j >> 1) * 32 + (j & 1) * 8);
          At[s * 8 + 2 * j + 0] = t.x;
          At[s * 8 + 2 * j + 1] = t.y;
        }

#pragma unroll
      for (int nt = 0; nt < 4; ++nt) {
        v8f z = v8f{};
        v8f p = __builtin_amdgcn_wmma_f32_16x16x128_fp8_fp8(
            At, Bt[nt], (short)0, z, false, false);
#pragma unroll
        for (int i = 0; i < 4; ++i) {
          acc[mt][nt][i]     += p[i]     * sc[mt][0][i];
          acc[mt][nt][i + 4] += p[i + 4] * sc[mt][1][i];
        }
      }
    }
  }

#pragma unroll
  for (int mt = 0; mt < 2; ++mt)
#pragma unroll
    for (int nt = 0; nt < 4; ++nt) {
      const int n = Nbase + nt * 16 + l16;
      const float wsv = wscale[n >> 7];
      const float bv  = bias[n];
#pragma unroll
      for (int i = 0; i < 8; ++i) {
        const int m = Mbase + mt * 16 + h * 8 + i;
        y[(size_t)m * N + n] = acc[mt][nt][i] * wsv + bv;
      }
    }
}

extern "C" void kernel_launch(void* const* d_in, const int* in_sizes, int n_in,
                              void* d_out, int out_size, void* d_ws, size_t ws_size,
                              hipStream_t stream) {
  const float* x    = (const float*)d_in[0];
  const float* w    = (const float*)d_in[1];
  const float* wsc  = (const float*)d_in[2];
  const float* bias = (const float*)d_in[3];
  float* y = (float*)d_out;

  const int N  = in_sizes[3];
  const int K  = (int)((long)in_sizes[1] / N);
  const int M  = (int)((long)in_sizes[0] / K);
  const int KB = K / QBLOCK;

  uint8_t* xq  = (uint8_t*)d_ws;
  uint8_t* wqp = xq + (size_t)M * K;
  float*   xst = (float*)(wqp + (size_t)N * K);

  {
    const long threads = (long)M * KB * 32;
    dim3 grid((unsigned)((threads + 255) / 256));
    quant_x_kernel<<<grid, 256, 0, stream>>>(x, xq, xst, M, K);
  }
  {
    const long n4 = (long)N * K / 4;
    dim3 grid((unsigned)((n4 + 255) / 256));
    quant_w_kernel<<<grid, 256, 0, stream>>>(w, wqp, n4);
  }
  {
    dim3 grid(N / 256, M / 64);
    fp8_gemm_kernel<<<grid, 256, 0, stream>>>(xq, xst, wqp, wsc, bias, y,
                                              M, N, K);
  }
}
